// NewtonNet_65420941853022
// MI455X (gfx1250) — matrix operand order repaired and checked by reading, not compiled, not run on hardware
//
#include <hip/hip_runtime.h>

#define BB   8
#define AA   256
#define NNB  48
#define FF   128
#define NBAS 20
#define LL   3
#define NAT  2048      /* BB*AA   */
#define NE   98304     /* NAT*NNB */
#define CUTR 5.0f
#define EPSR 1e-8f

typedef unsigned int   u32;
typedef unsigned short u16;
typedef __attribute__((ext_vector_type(8)))  float  v8f;
typedef __attribute__((ext_vector_type(8)))  u32    v8u;
typedef __attribute__((ext_vector_type(4)))  u32    v4u;
typedef __attribute__((ext_vector_type(2)))  u32    v2u;
typedef __attribute__((ext_vector_type(16))) __bf16 v16bf;

union ABu { v8u u; v16bf b; };
union BF4 { v2u u2; u16 h[4]; };

__device__ __forceinline__ u16 f2bf(float f) {
    u32 u = __float_as_uint(f);
    u32 r = (u + 0x7FFFu + ((u >> 16) & 1u)) >> 16;   // round-nearest-even
    return (u16)r;
}
__device__ __forceinline__ float bf2f(u16 h) {
    return __uint_as_float(((u32)h) << 16);
}
__device__ __forceinline__ float silu_f(float x) {
    return x / (1.0f + __expf(-x));
}

__device__ __forceinline__ v8f wmma_bf16(v8u a, v8u b, v8f c) {
    ABu A, B; A.u = a; B.u = b;
    return __builtin_amdgcn_wmma_f32_16x16x32_bf16(
        false, A.b, false, B.b, (short)0, c, false, false);
}

// A fragment (16x32 bf16) from row-major bf16 (u32 K-pair view).
// lane l: M = l&15, half = l>>4. u32 indices are contiguous per half:
// v=0..3 -> (k0>>1) + v + 4*half ; v=4..7 -> (k0>>1) + v + 4 + 4*half
// => exactly two b128 loads per lane.
__device__ __forceinline__ v8u load_a_frag_g(const u32* __restrict__ p,
                                             int ldu, int m0, int k0) {
    int lane = threadIdx.x & 31;
    int half = lane >> 4;
    const u32* row = p + (size_t)(m0 + (lane & 15)) * ldu + (k0 >> 1) + 4 * half;
    v4u lo = *(const v4u*)row;
    v4u hi = *(const v4u*)(row + 8);
    return __builtin_shufflevector(lo, hi, 0, 1, 2, 3, 4, 5, 6, 7);
}

// B fragment (32x16 bf16) from fragment-ordered packed weights:
// Wp[((kt*ntiles + nt)*32 + lane)*8 + v]  -> one 32B load per lane.
__device__ __forceinline__ v8u load_b_frag(const u32* __restrict__ wp,
                                           int ntiles, int kt, int nt) {
    int lane = threadIdx.x & 31;
    const u32* p = wp + ((size_t)((kt * ntiles + nt) * 32 + lane) << 3);
    return *(const v8u*)p;
}

// ---------------------------------------------------------------------------
// Fused 2-layer MLP:  out = act2( silu(A @ W1 + b1) @ W2 + b2 )
// A: [M][128] bf16 row-major. W1/W2: fragment-ordered packed bf16.
// 256 threads = 8 waves; each wave owns 16 rows; hidden staged in LDS.
// ---------------------------------------------------------------------------
template <int NT2, bool ACT2, bool OUTF, bool OUTB, bool HB1, bool HB2>
__global__ __launch_bounds__(256)
void mlp2_wmma(const u32* __restrict__ Abf,
               const u32* __restrict__ W1p, const float* __restrict__ b1,
               const u32* __restrict__ W2p, const float* __restrict__ b2,
               float* __restrict__ outF, u16* __restrict__ outB) {
    __shared__ u16 hid[128 * 128];
    const int lane = threadIdx.x & 31;
    const int wave = threadIdx.x >> 5;
    const int half = lane >> 4;
    const int mBlk = blockIdx.x * 128;
    const int mW   = wave * 16;

    // prefetch stage-2 weights while stage 1 runs (global_prefetch_b8)
    __builtin_prefetch(W2p + ((size_t)threadIdx.x << 3), 0, 1);

    // ---- stage 1: hidden = silu(A @ W1 + b1) -> LDS ----
    v8u afr[4];
#pragma unroll
    for (int kt = 0; kt < 4; ++kt)
        afr[kt] = load_a_frag_g(Abf, 64, mBlk + mW, kt * 32);

#pragma unroll
    for (int nt = 0; nt < 8; ++nt) {
        v8f acc = {};
#pragma unroll
        for (int kt = 0; kt < 4; ++kt)
            acc = wmma_bf16(afr[kt], load_b_frag(W1p, 8, kt, nt), acc);
        float bias = HB1 ? b1[nt * 16 + (lane & 15)] : 0.0f;
        int col = nt * 16 + (lane & 15);
#pragma unroll
        for (int r = 0; r < 8; ++r) {
            int m = mW + r + 8 * half;
            hid[m * 128 + col] = f2bf(silu_f(acc[r] + bias));
        }
    }
    __syncthreads();

    // ---- stage 2: out = act2(hidden @ W2 + b2) ----
    v8u hfr[4];
    {
        const u32* hrow = (const u32*)&hid[(mW + (lane & 15)) * 128];
#pragma unroll
        for (int kt = 0; kt < 4; ++kt) {
            const u32* q = hrow + kt * 16 + 4 * half;
            v4u lo = *(const v4u*)q;
            v4u hi = *(const v4u*)(q + 8);
            hfr[kt] = __builtin_shufflevector(lo, hi, 0, 1, 2, 3, 4, 5, 6, 7);
        }
    }
#pragma unroll
    for (int nt = 0; nt < NT2; ++nt) {
        v8f acc = {};
#pragma unroll
        for (int kt = 0; kt < 4; ++kt)
            acc = wmma_bf16(hfr[kt], load_b_frag(W2p, NT2, kt, nt), acc);
        float bias = HB2 ? b2[nt * 16 + (lane & 15)] : 0.0f;
        int col = nt * 16 + (lane & 15);
#pragma unroll
        for (int r = 0; r < 8; ++r) {
            int m = mBlk + mW + r + 8 * half;
            float v = acc[r] + bias;
            if (ACT2) v = silu_f(v);
            if (OUTF) outF[(size_t)m * (NT2 * 16) + col] = v;
            if (OUTB) outB[(size_t)m * (NT2 * 16) + col] = f2bf(v);
        }
    }
}

// ---------------------------------------------------------------------------
// m_e = (rbf @ W_ime + b_ime) * cut   (K padded 20->32, single WMMA K-step)
// ---------------------------------------------------------------------------
__global__ __launch_bounds__(256)
void me_gemm(const u32* __restrict__ rbfp,   // [NE][16] u32 (32 bf16, padded)
             const u32* __restrict__ Wimep,  // fragment-packed [1][8]
             const float* __restrict__ bime,
             const float* __restrict__ cutv,
             u16* __restrict__ meB) {
    const int lane = threadIdx.x & 31;
    const int wave = threadIdx.x >> 5;
    const int half = lane >> 4;
    const int m0 = blockIdx.x * 128 + wave * 16;

    v8u afr = load_a_frag_g(rbfp, 16, m0, 0);
    float cutr[8];
#pragma unroll
    for (int r = 0; r < 8; ++r) cutr[r] = cutv[m0 + r + 8 * half];

#pragma unroll
    for (int nt = 0; nt < 8; ++nt) {
        v8f acc = {};
        acc = wmma_bf16(afr, load_b_frag(Wimep, 8, 0, nt), acc);
        float bias = bime[nt * 16 + (lane & 15)];
        int col = nt * 16 + (lane & 15);
#pragma unroll
        for (int r = 0; r < 8; ++r) {
            int m = m0 + r + 8 * half;
            meB[(size_t)m * 128 + col] = f2bf((acc[r] + bias) * cutr[r]);
        }
    }
}

// ---------------------------------------------------------------------------
// Weight pack: row-major f32 [K][N] -> fragment-ordered bf16 K-pairs.
// Wp[((kt*NT + nt)*32 + lane)*8 + v] = {W[k][n], W[k+1][n]} with
// k = kt*32 + 2v + 8*(lane>>4) + 8*(v>>2), n = nt*16 + (lane&15).
// ---------------------------------------------------------------------------
__global__ void pack_w(const float* __restrict__ W, u32* __restrict__ Wp,
                       int K, int N, int Kp) {
    int idx = blockIdx.x * blockDim.x + threadIdx.x;
    int tot = (Kp >> 1) * N;
    if (idx >= tot) return;
    int NT = N >> 4;
    int v    = idx & 7;
    int lane = (idx >> 3) & 31;
    int fr   = idx >> 8;           // kt*NT + nt
    int nt = fr % NT;
    int kt = fr / NT;
    int n = nt * 16 + (lane & 15);
    int k = kt * 32 + 2 * v + 8 * (lane >> 4) + 8 * (v >> 2);
    float f0 = (k     < K) ? W[(size_t)k * N + n]       : 0.0f;
    float f1 = (k + 1 < K) ? W[(size_t)(k + 1) * N + n] : 0.0f;
    Wp[idx] = (u32)f2bf(f0) | ((u32)f2bf(f1) << 16);
}

// ---------------------------------------------------------------------------
// prep: a = embed[Z]; zero Fn / fn / dr0.  One wave per atom, 4 feat/lane.
// ---------------------------------------------------------------------------
__global__ __launch_bounds__(32)
void prep(const int* __restrict__ Z, const float* __restrict__ embed,
          float* __restrict__ aF, u16* __restrict__ aB,
          float* __restrict__ FnF, float* __restrict__ fnF,
          float* __restrict__ dr0) {
    int i  = blockIdx.x;
    int f0 = (threadIdx.x & 31) * 4;
    int z = Z[i];
    float4 v = *(const float4*)(embed + (size_t)z * FF + f0);
    *(float4*)(aF + (size_t)i * FF + f0) = v;
    BF4 p;
    p.h[0] = f2bf(v.x); p.h[1] = f2bf(v.y); p.h[2] = f2bf(v.z); p.h[3] = f2bf(v.w);
    *(v2u*)(aB + (size_t)i * FF + f0) = p.u2;
    float4 zf = make_float4(0.f, 0.f, 0.f, 0.f);
#pragma unroll
    for (int x = 0; x < 3; ++x) {
        *(float4*)(fnF + ((size_t)i * 3 + x) * FF + f0) = zf;
        *(float4*)(dr0 + ((size_t)i * 3 + x) * FF + f0) = zf;
    }
    if (threadIdx.x < 3) FnF[i * 3 + threadIdx.x] = 0.0f;
}

// ---------------------------------------------------------------------------
// geometry: dvec, cutoff, bessel RBF (padded to 32, bf16, paired stores)
// ---------------------------------------------------------------------------
__global__ __launch_bounds__(256)
void geom(const float* __restrict__ R, const int* __restrict__ nbr,
          u16* __restrict__ rbfB, float* __restrict__ dvecF,
          float* __restrict__ cutF) {
    int e = blockIdx.x * blockDim.x + threadIdx.x;
    if (e >= NE) return;
    int i = e / NNB;
    int b = i / AA;
    int j = nbr[e];
    const float* Ri = R + (size_t)i * 3;
    const float* Rj = R + ((size_t)b * AA + j) * 3;
    float dx = Rj[0] - Ri[0], dy = Rj[1] - Ri[1], dz = Rj[2] - Ri[2];
    float d = sqrtf(dx * dx + dy * dy + dz * dz);
    float inv = 1.0f / (d + EPSR);
    dvecF[(size_t)e * 3 + 0] = dx * inv;
    dvecF[(size_t)e * 3 + 1] = dy * inv;
    dvecF[(size_t)e * 3 + 2] = dz * inv;
    float xr = d / CUTR;
    float x2 = xr * xr;
    float fc = 1.0f - 6.0f * x2 * x2 * xr + 15.0f * x2 * x2 - 10.0f * x2 * xr;
    cutF[e] = (xr < 1.0f) ? fc : 0.0f;
    float coef = sqrtf(2.0f / CUTR);
    float w = 3.14159265358979323846f * d / CUTR;
#pragma unroll
    for (int n4 = 0; n4 < 8; ++n4) {
        BF4 p;
#pragma unroll
        for (int k = 0; k < 4; ++k) {
            int n = n4 * 4 + k;
            float rv = (n < NBAS) ? coef * __sinf((float)(n + 1) * w) * inv : 0.0f;
            p.h[k] = f2bf(rv);
        }
        *(v2u*)(rbfB + (size_t)e * 32 + n4 * 4) = p.u2;
    }
}

// ---------------------------------------------------------------------------
// msg = m_e * phi[nbr] * phi[self]; a += sum_nn msg*m; Fc = msg . Wc;
// F_msg = Fc * dvec; Fn += sum F_msg*m.
// One wave per atom, 4 features per lane, shfl_xor reduction (no barriers).
// ---------------------------------------------------------------------------
__global__ __launch_bounds__(256)
void msg_kernel(const u16* __restrict__ meB, const u16* __restrict__ phiB,
                const int* __restrict__ nbr, const float* __restrict__ nmask,
                const float* __restrict__ Wcl, const float* __restrict__ dvecF,
                u16* __restrict__ msgB, float* __restrict__ FmsgF,
                float* __restrict__ aF, u16* __restrict__ aB,
                float* __restrict__ FnF) {
    int lane = threadIdx.x & 31;
    int i = blockIdx.x * 8 + (threadIdx.x >> 5);
    int b = i / AA;
    int f0 = lane * 4;
    BF4 ps; ps.u2 = *(const v2u*)(phiB + (size_t)i * FF + f0);
    float phis[4];
#pragma unroll
    for (int k = 0; k < 4; ++k) phis[k] = bf2f(ps.h[k]);
    float4 wcv = *(const float4*)(Wcl + f0);
    float wck[4] = {wcv.x, wcv.y, wcv.z, wcv.w};
    float a_acc[4] = {0.f, 0.f, 0.f, 0.f};
    float fn_acc = 0.0f;
    for (int nn = 0; nn < NNB; ++nn) {
        int e = i * NNB + nn;
        int j = nbr[e];
        float m = nmask[e];
        BF4 me4; me4.u2 = *(const v2u*)(meB + (size_t)e * FF + f0);
        BF4 pn4; pn4.u2 = *(const v2u*)(phiB + ((size_t)b * AA + j) * FF + f0);
        BF4 st;
        float fc = 0.0f;
#pragma unroll
        for (int k = 0; k < 4; ++k) {
            float msg = bf2f(me4.h[k]) * bf2f(pn4.h[k]) * phis[k];
            st.h[k] = f2bf(msg);
            a_acc[k] += msg * m;
            fc += msg * wck[k];
        }
        *(v2u*)(msgB + (size_t)e * FF + f0) = st.u2;
#pragma unroll
        for (int s = 16; s > 0; s >>= 1) fc += __shfl_xor(fc, s, 32);
        if (lane < 3) {
            float fm = fc * dvecF[(size_t)e * 3 + lane];
            FmsgF[(size_t)e * 3 + lane] = fm;
            fn_acc += fm * m;
        }
    }
    float4 av = *(const float4*)(aF + (size_t)i * FF + f0);
    av.x += a_acc[0]; av.y += a_acc[1]; av.z += a_acc[2]; av.w += a_acc[3];
    *(float4*)(aF + (size_t)i * FF + f0) = av;
    BF4 ab;
    ab.h[0] = f2bf(av.x); ab.h[1] = f2bf(av.y);
    ab.h[2] = f2bf(av.z); ab.h[3] = f2bf(av.w);
    *(v2u*)(aB + (size_t)i * FF + f0) = ab.u2;
    if (lane < 3) FnF[i * 3 + lane] += fn_acc;
}

// ---------------------------------------------------------------------------
// upd_f[i,x,f] = sum_nn g[e,f]*F_msg[e,x]*m;  fn += upd_f
// ---------------------------------------------------------------------------
__global__ __launch_bounds__(256)
void updf_kernel(const u16* __restrict__ gB, const float* __restrict__ FmsgF,
                 const float* __restrict__ nmask,
                 float* __restrict__ updF, float* __restrict__ fnF) {
    int lane = threadIdx.x & 31;
    int i = blockIdx.x * 8 + (threadIdx.x >> 5);
    int f0 = lane * 4;
    float acc[3][4] = {};
    for (int nn = 0; nn < NNB; ++nn) {
        int e = i * NNB + nn;
        float m = nmask[e];
        BF4 g4; g4.u2 = *(const v2u*)(gB + (size_t)e * FF + f0);
        float fmx = FmsgF[(size_t)e * 3 + 0];
        float fmy = FmsgF[(size_t)e * 3 + 1];
        float fmz = FmsgF[(size_t)e * 3 + 2];
#pragma unroll
        for (int k = 0; k < 4; ++k) {
            float g = bf2f(g4.h[k]) * m;
            acc[0][k] += g * fmx;
            acc[1][k] += g * fmy;
            acc[2][k] += g * fmz;
        }
    }
    size_t base = (size_t)i * 3 * FF + f0;
#pragma unroll
    for (int x = 0; x < 3; ++x) {
        float4 u = make_float4(acc[x][0], acc[x][1], acc[x][2], acc[x][3]);
        *(float4*)(updF + base + x * FF) = u;
        float4 fo = *(const float4*)(fnF + base + x * FF);
        fo.x += u.x; fo.y += u.y; fo.z += u.z; fo.w += u.w;
        *(float4*)(fnF + base + x * FF) = fo;
    }
}

// ---------------------------------------------------------------------------
// drNew = drOld + sum_nn dr_edge[e,f]*drOld[nbr,x,f]*m + su[i,f]*upd_f[i,x,f]
// ---------------------------------------------------------------------------
__global__ __launch_bounds__(256)
void dr_kernel(const u16* __restrict__ deB, const int* __restrict__ nbr,
               const float* __restrict__ nmask, const float* __restrict__ drOld,
               const float* __restrict__ suF, const float* __restrict__ updF,
               float* __restrict__ drNew) {
    int lane = threadIdx.x & 31;
    int i = blockIdx.x * 8 + (threadIdx.x >> 5);
    int b = i / AA;
    int f0 = lane * 4;
    float acc[3][4] = {};
    for (int nn = 0; nn < NNB; ++nn) {
        int e = i * NNB + nn;
        float m = nmask[e];
        BF4 d4; d4.u2 = *(const v2u*)(deB + (size_t)e * FF + f0);
        size_t jb = ((size_t)b * AA + nbr[e]) * 3 * FF + f0;
#pragma unroll
        for (int x = 0; x < 3; ++x) {
            float4 dn = *(const float4*)(drOld + jb + x * FF);
            float de0 = bf2f(d4.h[0]) * m, de1 = bf2f(d4.h[1]) * m;
            float de2 = bf2f(d4.h[2]) * m, de3 = bf2f(d4.h[3]) * m;
            acc[x][0] += de0 * dn.x; acc[x][1] += de1 * dn.y;
            acc[x][2] += de2 * dn.z; acc[x][3] += de3 * dn.w;
        }
    }
    float4 su = *(const float4*)(suF + (size_t)i * FF + f0);
    size_t base = (size_t)i * 3 * FF + f0;
#pragma unroll
    for (int x = 0; x < 3; ++x) {
        float4 dro = *(const float4*)(drOld + base + x * FF);
        float4 uf  = *(const float4*)(updF + base + x * FF);
        float4 o;
        o.x = dro.x + acc[x][0] + su.x * uf.x;
        o.y = dro.y + acc[x][1] + su.y * uf.y;
        o.z = dro.z + acc[x][2] + su.z * uf.z;
        o.w = dro.w + acc[x][3] + su.w * uf.w;
        *(float4*)(drNew + base + x * FF) = o;
    }
}

// ---------------------------------------------------------------------------
// a += -is * sum_x fn[x]*dr[x]
// ---------------------------------------------------------------------------
__global__ __launch_bounds__(256)
void a2_kernel(const float* __restrict__ fnF, const float* __restrict__ drNew,
               const float* __restrict__ isF,
               float* __restrict__ aF, u16* __restrict__ aB) {
    int lane = threadIdx.x & 31;
    int i = blockIdx.x * 8 + (threadIdx.x >> 5);
    int f0 = lane * 4;
    size_t base = (size_t)i * 3 * FF + f0;
    float s[4] = {0.f, 0.f, 0.f, 0.f};
#pragma unroll
    for (int x = 0; x < 3; ++x) {
        float4 fn = *(const float4*)(fnF + base + x * FF);
        float4 dr = *(const float4*)(drNew + base + x * FF);
        s[0] += fn.x * dr.x; s[1] += fn.y * dr.y;
        s[2] += fn.z * dr.z; s[3] += fn.w * dr.w;
    }
    float4 is = *(const float4*)(isF + (size_t)i * FF + f0);
    float4 av = *(const float4*)(aF + (size_t)i * FF + f0);
    av.x -= is.x * s[0]; av.y -= is.y * s[1];
    av.z -= is.z * s[2]; av.w -= is.w * s[3];
    *(float4*)(aF + (size_t)i * FF + f0) = av;
    BF4 ab;
    ab.h[0] = f2bf(av.x); ab.h[1] = f2bf(av.y);
    ab.h[2] = f2bf(av.z); ab.h[3] = f2bf(av.w);
    *(v2u*)(aB + (size_t)i * FF + f0) = ab.u2;
}

// ---------------------------------------------------------------------------
// head tail: out[b] = sum_atoms mask * (h2 . Wh3 + bh3)
// ---------------------------------------------------------------------------
__global__ __launch_bounds__(256)
void head_final(const float* __restrict__ h2, const float* __restrict__ Wh3,
                const float* __restrict__ bh3, const float* __restrict__ amask,
                float* __restrict__ out) {
    __shared__ float red[256];
    int b = blockIdx.x;
    int t = threadIdx.x;                  // one atom per thread (A == 256)
    int i = b * AA + t;
    float acc = 0.0f;
#pragma unroll
    for (int k4 = 0; k4 < 16; ++k4) {
        float4 h = *(const float4*)(h2 + (size_t)i * 64 + k4 * 4);
        float4 w = *(const float4*)(Wh3 + k4 * 4);
        acc += h.x * w.x + h.y * w.y + h.z * w.z + h.w * w.w;
    }
    acc = (acc + bh3[0]) * amask[i];
    red[t] = acc;
    __syncthreads();
    for (int s = 128; s > 0; s >>= 1) {
        if (t < s) red[t] += red[t + s];
        __syncthreads();
    }
    if (t == 0) out[b] = red[0];
}

// ===========================================================================
extern "C" void kernel_launch(void* const* d_in, const int* in_sizes, int n_in,
                              void* d_out, int out_size, void* d_ws,
                              size_t ws_size, hipStream_t stream) {
    (void)in_sizes; (void)n_in; (void)out_size; (void)ws_size;

    const int*   Z     = (const int*)  d_in[0];
    const float* R     = (const float*)d_in[1];
    const int*   nbr   = (const int*)  d_in[2];
    const float* nmask = (const float*)d_in[3];
    const float* amask = (const float*)d_in[4];
    const float* embed = (const float*)d_in[5];
    const float* W_ime = (const float*)d_in[6];
    const float* b_ime = (const float*)d_in[7];
    const float* Wmn1  = (const float*)d_in[8];  const float* bmn1 = (const float*)d_in[9];
    const float* Wmn2  = (const float*)d_in[10]; const float* bmn2 = (const float*)d_in[11];
    const float* Wc    = (const float*)d_in[12];
    const float* Wmf1  = (const float*)d_in[13]; const float* bmf1 = (const float*)d_in[14];
    const float* Wmf2  = (const float*)d_in[15]; const float* bmf2 = (const float*)d_in[16];
    const float* Wsu1  = (const float*)d_in[17]; const float* bsu1 = (const float*)d_in[18];
    const float* Wsu2  = (const float*)d_in[19]; const float* bsu2 = (const float*)d_in[20];
    const float* Wme1  = (const float*)d_in[21]; const float* Wme2 = (const float*)d_in[22];
    const float* Wis1  = (const float*)d_in[23]; const float* bis1 = (const float*)d_in[24];
    const float* Wis2  = (const float*)d_in[25]; const float* bis2 = (const float*)d_in[26];
    const float* Wh1   = (const float*)d_in[27]; const float* bh1  = (const float*)d_in[28];
    const float* Wh2   = (const float*)d_in[29]; const float* bh2  = (const float*)d_in[30];
    const float* Wh3   = (const float*)d_in[31]; const float* bh3  = (const float*)d_in[32];
    float* out = (float*)d_out;

    char* wsb = (char*)d_ws;
    size_t off = 0;
    auto alloc = [&](size_t bytes) -> char* {
        char* p = wsb + off;
        off = (off + bytes + 255) & ~(size_t)255;
        return p;
    };

    float* aF    = (float*)alloc((size_t)NAT * FF * 4);
    u16*   aB    = (u16*)  alloc((size_t)NAT * FF * 2);
    u16*   phiB  = (u16*)  alloc((size_t)NAT * FF * 2);
    float* suF   = (float*)alloc((size_t)NAT * FF * 4);
    float* isF   = (float*)alloc((size_t)NAT * FF * 4);
    float* FnF   = (float*)alloc((size_t)NAT * 3 * 4);
    float* fnF   = (float*)alloc((size_t)NAT * 3 * FF * 4);
    float* dr0   = (float*)alloc((size_t)NAT * 3 * FF * 4);
    float* dr1   = (float*)alloc((size_t)NAT * 3 * FF * 4);
    float* updF  = (float*)alloc((size_t)NAT * 3 * FF * 4);
    u16*   rbfB  = (u16*)  alloc((size_t)NE * 32 * 2);
    float* dvecF = (float*)alloc((size_t)NE * 3 * 4);
    float* cutF  = (float*)alloc((size_t)NE * 4);
    u16*   meB   = (u16*)  alloc((size_t)NE * FF * 2);
    u16*   msgB  = (u16*)  alloc((size_t)NE * FF * 2);
    u16*   gB    = (u16*)  alloc((size_t)NE * FF * 2);
    u16*   deB   = (u16*)  alloc((size_t)NE * FF * 2);
    float* FmsgF = (float*)alloc((size_t)NE * 3 * 4);
    float* h2F   = (float*)alloc((size_t)NAT * 64 * 4);

    u32* pk[LL][10];
    for (int l = 0; l < LL; ++l)
        for (int w = 0; w < 10; ++w)
            pk[l][w] = (u32*)alloc((size_t)64 * 128 * 4);
    u32* pime[LL];
    for (int l = 0; l < LL; ++l) pime[l] = (u32*)alloc((size_t)16 * 128 * 4);
    u32* pWh1 = (u32*)alloc((size_t)64 * 128 * 4);
    u32* pWh2 = (u32*)alloc((size_t)64 * 64 * 4);

    // ---- pack weights to fragment-ordered bf16 layout ----
    const float* wsrc[10] = {Wmn1, Wmn2, Wmf1, Wmf2, Wsu1,
                             Wsu2, Wme1, Wme2, Wis1, Wis2};
    for (int l = 0; l < LL; ++l) {
        for (int w = 0; w < 10; ++w)
            pack_w<<<32, 256, 0, stream>>>(wsrc[w] + (size_t)l * FF * FF,
                                           pk[l][w], FF, FF, FF);
        pack_w<<<8, 256, 0, stream>>>(W_ime + (size_t)l * NBAS * FF,
                                      pime[l], NBAS, FF, 32);
    }
    pack_w<<<32, 256, 0, stream>>>(Wh1, pWh1, 128, 128, 128);
    pack_w<<<16, 256, 0, stream>>>(Wh2, pWh2, 128, 64, 128);

    // ---- init + geometry ----
    prep<<<NAT, 32, 0, stream>>>(Z, embed, aF, aB, FnF, fnF, dr0);
    geom<<<NE / 256, 256, 0, stream>>>(R, nbr, rbfB, dvecF, cutF);

    float* drA = dr0;
    float* drB = dr1;
    for (int l = 0; l < LL; ++l) {
        // phi = mlp2(a; Wmn)           -> bf16
        mlp2_wmma<8, false, false, true, true, true>
            <<<NAT / 128, 256, 0, stream>>>((const u32*)aB, pk[l][0],
                bmn1 + (size_t)l * FF, pk[l][1], bmn2 + (size_t)l * FF,
                nullptr, phiB);
        // m_e = (rbf @ W_ime + b) * cut
        me_gemm<<<NE / 128, 256, 0, stream>>>((const u32*)rbfB, pime[l],
                b_ime + (size_t)l * FF, cutF, meB);
        // msg + a update + F_msg + Fn
        msg_kernel<<<NAT / 8, 256, 0, stream>>>(meB, phiB, nbr, nmask,
                Wc + (size_t)l * FF, dvecF, msgB, FmsgF, aF, aB, FnF);
        // g = mlp2(msg; Wmf)           -> bf16
        mlp2_wmma<8, false, false, true, true, true>
            <<<NE / 128, 256, 0, stream>>>((const u32*)msgB, pk[l][2],
                bmf1 + (size_t)l * FF, pk[l][3], bmf2 + (size_t)l * FF,
                nullptr, gB);
        // upd_f + fn
        updf_kernel<<<NAT / 8, 256, 0, stream>>>(gB, FmsgF, nmask, updF, fnF);
        // dr_edge = silu(msg @ Wme1) @ Wme2   (no biases) -> bf16
        mlp2_wmma<8, false, false, true, false, false>
            <<<NE / 128, 256, 0, stream>>>((const u32*)msgB, pk[l][6],
                nullptr, pk[l][7], nullptr, nullptr, deB);
        // su = mlp2(a; Wsu)            -> f32
        mlp2_wmma<8, false, true, false, true, true>
            <<<NAT / 128, 256, 0, stream>>>((const u32*)aB, pk[l][4],
                bsu1 + (size_t)l * FF, pk[l][5], bsu2 + (size_t)l * FF,
                suF, nullptr);
        // dr recursion (double-buffered)
        dr_kernel<<<NAT / 8, 256, 0, stream>>>(deB, nbr, nmask, drA, suF,
                                               updF, drB);
        // is = mlp2(a; Wis)            -> f32
        mlp2_wmma<8, false, true, false, true, true>
            <<<NAT / 128, 256, 0, stream>>>((const u32*)aB, pk[l][8],
                bis1 + (size_t)l * FF, pk[l][9], bis2 + (size_t)l * FF,
                isF, nullptr);
        // a += -is * sum(fn * dr)
        a2_kernel<<<NAT / 8, 256, 0, stream>>>(fnF, drB, isF, aF, aB);

        float* t = drA; drA = drB; drB = t;
    }

    // head: h2 = silu(silu(a@Wh1+b1)@Wh2+b2)   (N2 = 64)
    mlp2_wmma<4, true, true, false, true, true>
        <<<NAT / 128, 256, 0, stream>>>((const u32*)aB, pWh1, bh1, pWh2, bh2,
                                        h2F, nullptr);
    head_final<<<BB, 256, 0, stream>>>(h2F, Wh3, bh3, amask, out);
}
